// HDRNet_41377714929736
// MI455X (gfx1250) — matrix operand order repaired and checked by reading, not compiled, not run on hardware
//
#include <hip/hip_runtime.h>
#include <math.h>

typedef __attribute__((ext_vector_type(16))) _Float16 v16h;
typedef __attribute__((ext_vector_type(8)))  float    v8f;

#define BN_EPS 1e-5f
static inline int cdiv(int a, int b) { return (a + b - 1) / b; }

// ---------------------------------------------------------------------------
// Bilinear resize NCHW (768 -> 256), one thread per output element.
// ---------------------------------------------------------------------------
__global__ void k_resize(const float* __restrict__ x, float* __restrict__ out,
                         int BC, int Hin, int Win, int Hout, int Wout) {
  int idx = blockIdx.x * blockDim.x + threadIdx.x;
  int total = BC * Hout * Wout;
  if (idx >= total) return;
  int ow = idx % Wout;
  int oh = (idx / Wout) % Hout;
  int bc = idx / (Wout * Hout);
  float sy = (oh + 0.5f) * ((float)Hin / (float)Hout) - 0.5f;
  sy = fmaxf(sy, 0.0f);
  int y0 = min((int)floorf(sy), Hin - 1);
  int y1 = min(y0 + 1, Hin - 1);
  float ty = sy - (float)y0;
  float sx = (ow + 0.5f) * ((float)Win / (float)Wout) - 0.5f;
  sx = fmaxf(sx, 0.0f);
  int x0 = min((int)floorf(sx), Win - 1);
  int x1 = min(x0 + 1, Win - 1);
  float tx = sx - (float)x0;
  const float* p = x + (size_t)bc * Hin * Win;
  float v00 = p[y0 * Win + x0], v01 = p[y0 * Win + x1];
  float v10 = p[y1 * Win + x0], v11 = p[y1 * Win + x1];
  float v0 = v00 + (v01 - v00) * tx;
  float v1 = v10 + (v11 - v10) * tx;
  out[idx] = v0 + (v1 - v0) * ty;
}

// ---------------------------------------------------------------------------
// Direct conv3x3 stride-s pad-1 + BN + ReLU (used only for ll1, Cin=3).
// ---------------------------------------------------------------------------
__global__ void k_conv_direct(const float* __restrict__ in, const float* __restrict__ wgt,
                              const float* __restrict__ g, const float* __restrict__ bb,
                              const float* __restrict__ mm, const float* __restrict__ vv,
                              float* __restrict__ out,
                              int B_, int Cin, int Cout, int Hin, int Win,
                              int Hout, int Wout, int stride) {
  int idx = blockIdx.x * blockDim.x + threadIdx.x;
  int total = B_ * Cout * Hout * Wout;
  if (idx >= total) return;
  int ow = idx % Wout;
  int oh = (idx / Wout) % Hout;
  int co = (idx / (Wout * Hout)) % Cout;
  int b  = idx / (Wout * Hout * Cout);
  float acc = 0.0f;
  for (int ci = 0; ci < Cin; ++ci) {
    const float* ip = in + (((size_t)b * Cin + ci) * Hin) * Win;
    const float* wp = wgt + ((size_t)co * Cin + ci) * 9;
#pragma unroll
    for (int kh = 0; kh < 3; ++kh) {
      int ih = oh * stride - 1 + kh;
      if (ih < 0 || ih >= Hin) continue;
#pragma unroll
      for (int kw = 0; kw < 3; ++kw) {
        int iw = ow * stride - 1 + kw;
        if (iw < 0 || iw >= Win) continue;
        acc += ip[ih * Win + iw] * wp[kh * 3 + kw];
      }
    }
  }
  float inv = g[co] * rsqrtf(vv[co] + BN_EPS);
  float y = acc * inv + (bb[co] - mm[co] * inv);
  out[idx] = fmaxf(y, 0.0f);
}

// ---------------------------------------------------------------------------
// Implicit-GEMM conv via V_WMMA_F32_16X16X32_F16 (wave32; one wave per block).
//   GEMM: M = Cout (16-tile), N = 16 output pixels, K = Cin*KS*KS (pad to 32).
//   KS is a template parameter so the k -> (ci,kh,kw) decomposition uses
//   compile-time magic-multiply instead of runtime integer division.
//   BN scale folded into A (weights); BN shift / bias + ReLU in epilogue.
// A layout (16x32 f16): lane = m + 16*hi ; elem e -> K = (e&7) + 8*hi + 16*(e>>3)
// B layout (32x16 f16): lane = N + 16*hi ; elem e -> K = (e&7) + 8*hi + 16*(e>>3)
// C/D layout (16x16 f32): N = lane&15 ; row = r + 8*(lane>>4)
// ---------------------------------------------------------------------------
template <int KS>
__global__ __launch_bounds__(32) void k_conv_wmma(
    const float* __restrict__ in, const float* __restrict__ wgt,
    const float* __restrict__ bng, const float* __restrict__ bnb,
    const float* __restrict__ bnm, const float* __restrict__ bnv,
    const float* __restrict__ bias, float* __restrict__ out,
    int Cin, int Cout, int Hin, int Win, int Hout, int Wout,
    int stride, int pad, int use_bn, int use_relu) {
  const int lane = threadIdx.x;      // 0..31, EXEC all ones
  const int hi   = lane >> 4;
  const int l16  = lane & 15;
  const int b      = blockIdx.z;
  const int m_base = blockIdx.y * 16;
  const int n_base = blockIdx.x * 16;
  constexpr int ks2 = KS * KS;
  const int K    = Cin * ks2;
  const int Kpad = (K + 31) & ~31;
  const int npix = Hout * Wout;

  const int mA = m_base + l16;                       // A-fragment row (out channel)
  float inv = 1.0f;
  if (use_bn && mA < Cout) inv = bng[mA] * rsqrtf(bnv[mA] + BN_EPS);

  const int nB = n_base + l16;                       // B-fragment column (pixel)
  const int oh = (nB < npix) ? (nB / Wout) : 0;
  const int ow = (nB < npix) ? (nB % Wout) : 0;
  const int ihb = oh * stride - pad;                 // input tap base
  const int iwb = ow * stride - pad;
  const float* inb = in + (size_t)b * Cin * Hin * Win;
  const float* wb  = wgt + (size_t)(mA < Cout ? mA : 0) * K;

  v8f acc = {};
  for (int kc = 0; kc < Kpad; kc += 32) {
    v16h afrag, bfrag;
#pragma unroll
    for (int e = 0; e < 16; ++e) {
      // ---- A element ----
      int ka = kc + (e & 7) + (hi << 3) + ((e >> 3) << 4);
      float av = 0.0f;
      if (mA < Cout && ka < K) av = wb[ka] * inv;
      afrag[e] = (_Float16)av;
      // ---- B element ----
      int kb = kc + (e & 7) + (hi << 3) + ((e >> 3) << 4);
      float bv = 0.0f;
      if (nB < npix && kb < K) {
        int ci = kb / ks2;       // compile-time constant divisor
        int r  = kb % ks2;
        int ih = ihb + (r / KS);
        int iw = iwb + (r % KS);
        if (ih >= 0 && ih < Hin && iw >= 0 && iw < Win)
          bv = inb[((size_t)ci * Hin + ih) * Win + iw];
      }
      bfrag[e] = (_Float16)bv;
    }
    acc = __builtin_amdgcn_wmma_f32_16x16x32_f16(false, afrag, false, bfrag,
                                                 (short)0, acc, false, false);
  }

#pragma unroll
  for (int r = 0; r < 8; ++r) {
    int mM = m_base + r + (hi << 3);
    if (mM >= Cout || nB >= npix) continue;
    float beta;
    if (use_bn) {
      float invm = bng[mM] * rsqrtf(bnv[mM] + BN_EPS);
      beta = bnb[mM] - bnm[mM] * invm;
    } else {
      beta = bias ? bias[mM] : 0.0f;
    }
    float y = acc[r] + beta;
    if (use_relu) y = fmaxf(y, 0.0f);
    out[(((size_t)b * Cout + mM) * Hout + oh) * Wout + ow] = y;
  }
}

// ---------------------------------------------------------------------------
// Global-average-pool over 16x16 + fc1(64->32,relu) + fc2(32->64,relu).
// One block, 128 threads, B=2.
// ---------------------------------------------------------------------------
__global__ void k_gap_fc(const float* __restrict__ f,   // (2,64,16,16)
                         const float* __restrict__ fc1w, const float* __restrict__ fc1b,
                         const float* __restrict__ fc2w, const float* __restrict__ fc2b,
                         float* __restrict__ g2out) {   // (2,64)
  __shared__ float gap[2 * 64];
  __shared__ float g1[2 * 32];
  int t = threadIdx.x;                      // 0..127
  {
    int b = t >> 6, c = t & 63;
    const float4* p = (const float4*)(f + ((size_t)(b * 64 + c)) * 256);
    float s = 0.0f;
    for (int i = 0; i < 64; ++i) {
      float4 v = p[i];
      s += v.x + v.y + v.z + v.w;
    }
    gap[b * 64 + c] = s * (1.0f / 256.0f);
  }
  __syncthreads();
  if (t < 64) {
    int b = t >> 5, j = t & 31;
    float s = fc1b[j];
    for (int c = 0; c < 64; ++c) s += fc1w[j * 64 + c] * gap[b * 64 + c];
    g1[b * 32 + j] = fmaxf(s, 0.0f);
  }
  __syncthreads();
  {
    int b = t >> 6, j = t & 63;
    float s = fc2b[j];
    for (int i = 0; i < 32; ++i) s += fc2w[j * 32 + i] * g1[b * 32 + i];
    g2out[b * 64 + j] = fmaxf(s, 0.0f);
  }
}

// high = local + g2[:, :, None, None]   (2,64,8,8)
__global__ void k_add_g2(const float* __restrict__ local, const float* __restrict__ g2,
                         float* __restrict__ high) {
  int idx = blockIdx.x * blockDim.x + threadIdx.x;
  if (idx >= 2 * 64 * 64) return;
  int c = (idx >> 6) & 63;
  int b = idx >> 12;
  high[idx] = local[idx] + g2[b * 64 + c];
}

// ---------------------------------------------------------------------------
// Fused full-res stage: guide (1x1 convs + sigmoid) + trilinear grid slice +
// per-pixel affine apply + clip. Grid (12,8,8,8) per image staged in LDS
// with b128 loads/stores.
// ---------------------------------------------------------------------------
__global__ void k_apply(const float* __restrict__ x,    // (2,3,768,768)
                        const float* __restrict__ raw,  // (2,96,8,8) == (2,12,8,8,8)
                        const float* __restrict__ gd1w, const float* __restrict__ gd1b,
                        const float* __restrict__ gd2w, const float* __restrict__ gd2b,
                        float* __restrict__ out, int H, int W) {
  __shared__ __align__(16) float sgrid[12 * 8 * 8 * 8];   // 24 KB
  const int b = blockIdx.y;
  {
    const float4* src = (const float4*)(raw + (size_t)b * 6144);
    float4* dst = (float4*)sgrid;
    for (int i = threadIdx.x; i < 1536; i += blockDim.x) dst[i] = src[i];
  }
  __syncthreads();

  int p = blockIdx.x * blockDim.x + threadIdx.x;
  int HW = H * W;
  if (p >= HW) return;
  int h = p / W, w = p % W;
  size_t base = (size_t)b * 3 * HW + p;
  float xr = x[base], xg = x[base + HW], xb = x[base + 2 * (size_t)HW];

  // ---- guide ----
  float acc2 = gd2b[0];
#pragma unroll
  for (int c = 0; c < 16; ++c) {
    float v = gd1w[c * 3 + 0] * xr + gd1w[c * 3 + 1] * xg + gd1w[c * 3 + 2] * xb + gd1b[c];
    acc2 += gd2w[c] * fmaxf(v, 0.0f);
  }
  float guide = 1.0f / (1.0f + expf(-acc2));

  // ---- spatial bilinear coords on the 8x8 grid ----
  float sy = (h + 0.5f) * (8.0f / (float)H) - 0.5f; sy = fmaxf(sy, 0.0f);
  int y0 = min((int)sy, 7); int y1 = min(y0 + 1, 7); float ty = sy - (float)y0;
  float sx = (w + 0.5f) * (8.0f / (float)W) - 0.5f; sx = fmaxf(sx, 0.0f);
  int x0 = min((int)sx, 7); int x1 = min(x0 + 1, 7); float tx = sx - (float)x0;
  float w00 = (1.f - ty) * (1.f - tx), w01 = (1.f - ty) * tx;
  float w10 = ty * (1.f - tx),         w11 = ty * tx;

  // ---- depth lerp coords ----
  float gs = guide * 7.0f;
  int dlo = min(max((int)gs, 0), 6);
  float alpha = gs - (float)dlo;

  float coeff[12];
#pragma unroll
  for (int c = 0; c < 12; ++c) {
    int b0 = (c * 8 + dlo) * 64;
    int b1 = b0 + 64;
    float lo = w00 * sgrid[b0 + y0 * 8 + x0] + w01 * sgrid[b0 + y0 * 8 + x1]
             + w10 * sgrid[b0 + y1 * 8 + x0] + w11 * sgrid[b0 + y1 * 8 + x1];
    float hv = w00 * sgrid[b1 + y0 * 8 + x0] + w01 * sgrid[b1 + y0 * 8 + x1]
             + w10 * sgrid[b1 + y1 * 8 + x0] + w11 * sgrid[b1 + y1 * 8 + x1];
    coeff[c] = lo + (hv - lo) * alpha;
  }

  float orr = coeff[0] * xr + coeff[1] * xg + coeff[2] * xb + coeff[9];
  float og  = coeff[3] * xr + coeff[4] * xg + coeff[5] * xb + coeff[10];
  float ob  = coeff[6] * xr + coeff[7] * xg + coeff[8] * xb + coeff[11];
  out[base]                    = fminf(fmaxf(orr, 0.0f), 1.0f);
  out[base + HW]               = fminf(fmaxf(og,  0.0f), 1.0f);
  out[base + 2 * (size_t)HW]   = fminf(fmaxf(ob,  0.0f), 1.0f);
}

// ---------------------------------------------------------------------------
extern "C" void kernel_launch(void* const* d_in, const int* in_sizes, int n_in,
                              void* d_out, int out_size, void* d_ws, size_t ws_size,
                              hipStream_t stream) {
  (void)in_sizes; (void)n_in; (void)out_size; (void)ws_size;
  const float* x = (const float*)d_in[0];
  // conv-block params: ll1,ll2,ll3,ll4,hl1,hl2,gr1 each {w,g,b,m,v} starting at 1
  const float* cb[7][5];
  for (int l = 0; l < 7; ++l)
    for (int j = 0; j < 5; ++j) cb[l][j] = (const float*)d_in[1 + l * 5 + j];
  const float* fc1w = (const float*)d_in[36];
  const float* fc1b = (const float*)d_in[37];
  const float* fc2w = (const float*)d_in[38];
  const float* fc2b = (const float*)d_in[39];
  const float* gr2w = (const float*)d_in[40];
  const float* gr2b = (const float*)d_in[41];
  const float* gd1w = (const float*)d_in[42];
  const float* gd1b = (const float*)d_in[43];
  const float* gd2w = (const float*)d_in[44];
  const float* gd2b = (const float*)d_in[45];
  float* out = (float*)d_out;

  float* ws = (float*)d_ws;
  float* lowres = ws;                  // 2*3*256*256 = 393216
  float* ll1o   = lowres + 393216;     // 2*8*128*128 = 262144
  float* ll2o   = ll1o + 262144;       // 2*16*64*64  = 131072
  float* ll3o   = ll2o + 131072;       // 2*32*32*32  =  65536
  float* ll4o   = ll3o + 65536;        // 2*64*16*16  =  32768  (== f)
  float* hl1o   = ll4o + 32768;        // 2*64*8*8    =   8192
  float* hl2o   = hl1o + 8192;         //               8192
  float* higho  = hl2o + 8192;         //               8192
  float* gr1o   = higho + 8192;        //               8192
  float* rawo   = gr1o + 8192;         // 2*96*8*8    =  12288
  float* g2o    = rawo + 12288;        // 2*64        =    128

  // 1) lowres = resize(x, 256, 256)
  {
    int total = 2 * 3 * 256 * 256;
    k_resize<<<cdiv(total, 256), 256, 0, stream>>>(x, lowres, 6, 768, 768, 256, 256);
  }
  // 2) ll1: 3->8, s2 : (2,8,128,128)  (direct conv; Cin too small for WMMA)
  {
    int total = 2 * 8 * 128 * 128;
    k_conv_direct<<<cdiv(total, 256), 256, 0, stream>>>(
        lowres, cb[0][0], cb[0][1], cb[0][2], cb[0][3], cb[0][4], ll1o,
        2, 3, 8, 256, 256, 128, 128, 2);
  }
  // 3) ll2: 8->16, s2 : (2,16,64,64)
  k_conv_wmma<3><<<dim3(cdiv(64 * 64, 16), 1, 2), 32, 0, stream>>>(
      ll1o, cb[1][0], cb[1][1], cb[1][2], cb[1][3], cb[1][4], nullptr, ll2o,
      8, 16, 128, 128, 64, 64, 2, 1, 1, 1);
  // 4) ll3: 16->32, s2 : (2,32,32,32)
  k_conv_wmma<3><<<dim3(cdiv(32 * 32, 16), 2, 2), 32, 0, stream>>>(
      ll2o, cb[2][0], cb[2][1], cb[2][2], cb[2][3], cb[2][4], nullptr, ll3o,
      16, 32, 64, 64, 32, 32, 2, 1, 1, 1);
  // 5) ll4: 32->64, s2 : (2,64,16,16)
  k_conv_wmma<3><<<dim3(cdiv(16 * 16, 16), 4, 2), 32, 0, stream>>>(
      ll3o, cb[3][0], cb[3][1], cb[3][2], cb[3][3], cb[3][4], nullptr, ll4o,
      32, 64, 32, 32, 16, 16, 2, 1, 1, 1);
  // 6) hl1: 64->64, s2 : (2,64,8,8)
  k_conv_wmma<3><<<dim3(cdiv(8 * 8, 16), 4, 2), 32, 0, stream>>>(
      ll4o, cb[4][0], cb[4][1], cb[4][2], cb[4][3], cb[4][4], nullptr, hl1o,
      64, 64, 16, 16, 8, 8, 2, 1, 1, 1);
  // 7) hl2: 64->64, s1 : (2,64,8,8)
  k_conv_wmma<3><<<dim3(cdiv(8 * 8, 16), 4, 2), 32, 0, stream>>>(
      hl1o, cb[5][0], cb[5][1], cb[5][2], cb[5][3], cb[5][4], nullptr, hl2o,
      64, 64, 8, 8, 8, 8, 1, 1, 1, 1);
  // 8) gap + fc1 + fc2 -> g2 (2,64)
  k_gap_fc<<<1, 128, 0, stream>>>(ll4o, fc1w, fc1b, fc2w, fc2b, g2o);
  // 9) high = hl2 + g2 broadcast
  k_add_g2<<<cdiv(2 * 64 * 64, 256), 256, 0, stream>>>(hl2o, g2o, higho);
  // 10) gr1: 64->64, s1 : (2,64,8,8)
  k_conv_wmma<3><<<dim3(cdiv(8 * 8, 16), 4, 2), 32, 0, stream>>>(
      higho, cb[6][0], cb[6][1], cb[6][2], cb[6][3], cb[6][4], nullptr, gr1o,
      64, 64, 8, 8, 8, 8, 1, 1, 1, 1);
  // 11) gr2: 1x1 conv 64->96 + bias (no BN, no relu) : raw (2,96,8,8)
  k_conv_wmma<1><<<dim3(cdiv(8 * 8, 16), 6, 2), 32, 0, stream>>>(
      gr1o, gr2w, nullptr, nullptr, nullptr, nullptr, gr2b, rawo,
      64, 96, 8, 8, 8, 8, 1, 0, 0, 0);
  // 12) fused guide + slice + apply over full res
  k_apply<<<dim3(cdiv(768 * 768, 256), 2), 256, 0, stream>>>(
      x, rawo, gd1w, gd1b, gd2w, gd2b, out, 768, 768);
}
